// MultiHeadAttention_89094801589003
// MI455X (gfx1250) — compile-verified, matrix operand order checked by reading, not run on hardware
//
#include <hip/hip_runtime.h>
#include <hip/hip_bf16.h>

// ---------------------------------------------------------------------------
// MI455X (gfx1250, wave32) fused multi-head attention layer.
// GEMMs: v_wmma_f32_16x16x32_bf16 (f32 accumulate).
// gemm_out A-tile staged with TENSOR_LOAD_TO_LDS (TDM) + s_wait_tensorcnt.
// Shapes (compile-time): B=4, S=2048, D=1024, H=16, DK=DV=64.
// ---------------------------------------------------------------------------

typedef __attribute__((ext_vector_type(16))) __bf16 v16bf;
typedef __attribute__((ext_vector_type(8)))  float  v8f;
typedef unsigned int u32;
typedef __attribute__((ext_vector_type(4))) u32 v4u;
typedef __attribute__((ext_vector_type(4))) int v4i;
typedef __attribute__((ext_vector_type(8))) int v8i;

#define CB 4
#define CS 2048
#define CD 1024
#define CH 16

__device__ __forceinline__ v8f wmma_bf16(v16bf a, v16bf b, v8f c) {
    return __builtin_amdgcn_wmma_f32_16x16x32_bf16(
        false, a, false, b, (short)0, c, false, false);
}

// 16x32 A-fragment (or 32x16 B-fragment from K-contiguous storage), per the
// documented bf16 VGPR layout: lanes<16 hold K[k0..k0+7]/K[k0+16..k0+23],
// lanes>=16 hold K[k0+8..k0+15]/K[k0+24..k0+31], row = M (A) or N (B).
__device__ __forceinline__ v16bf frag_ld(const __bf16* base, int stride,
                                         int row0, int k0, int lane) {
    const int half = lane >> 4, lr = lane & 15;
    const __bf16* p = base + (size_t)(row0 + lr) * stride;
    const int ka = k0 + half * 8;
    union { v16bf v; uint4 u[2]; } f;
    f.u[0] = *(const uint4*)(p + ka);
    f.u[1] = *(const uint4*)(p + ka + 16);
    return f.v;
}

__device__ __forceinline__ v16bf frag_ld_f32(const float* base, int stride,
                                             int row0, int k0, int lane) {
    const int half = lane >> 4, lr = lane & 15;
    const float* p = base + (size_t)(row0 + lr) * stride;
    const int ka = k0 + half * 8;
    union { v16bf v; __bf16 e[16]; } f;
#pragma unroll
    for (int j = 0; j < 8; ++j) {
        f.e[j]     = (__bf16)p[ka + j];
        f.e[8 + j] = (__bf16)p[ka + 16 + j];
    }
    return f.v;
}

__device__ __forceinline__ float wave_max32(float v) {
#pragma unroll
    for (int o = 16; o > 0; o >>= 1) v = fmaxf(v, __shfl_xor(v, o, 32));
    return v;
}
__device__ __forceinline__ float wave_sum32(float v) {
#pragma unroll
    for (int o = 16; o > 0; o >>= 1) v += __shfl_xor(v, o, 32);
    return v;
}

// ---------------------------------------------------------------------------
// TDM: load a 32x128 bf16 tile (128 rows x 32 elems each) from a row-major
// [8192 x 1024] bf16 tensor into LDS with padded row stride 40 elements.
// D# built per CDNA5 ISA ch.8: data_size=2B, pad_interval=16 DW (code 3),
// pad_amount=4 DW (code 3) -> LDS row stride 32+8 = 40 bf16 = 80 B.
// ---------------------------------------------------------------------------
__device__ __forceinline__ void tdm_load_tile(u32 lds_off, const __bf16* gtile) {
    unsigned long long ga = (unsigned long long)(uintptr_t)gtile;
    v4u g0;
    g0[0] = 1u;                                   // count=1, user mode
    g0[1] = lds_off;                              // lds_addr
    g0[2] = (u32)ga;                              // global_addr[31:0]
    g0[3] = (u32)((ga >> 32) & 0x01FFFFFFull) | (2u << 30);  // [56:32]+type=2
    v8i g1;
    g1[0] = (int)((1u << 16)       // data_size = 2 bytes
                | (1u << 20)       // pad_enable
                | (3u << 22)       // pad_interval: 16 DWORDs
                | (3u << 25));     // pad_amount: 4 DWORDs
    g1[1] = (int)(1024u << 16);    // tensor_dim0[15:0] into bits 63:48
    g1[2] = (int)(8192u << 16);    // dim0[31:16]=0 | tensor_dim1[15:0]
    g1[3] = (int)(32u << 16);      // tensor_dim1[31:16]=0 | tile_dim0=32
    g1[4] = 128;                   // tile_dim1=128 | tile_dim2=0
    g1[5] = 1024;                  // tensor_dim0_stride[31:0]
    g1[6] = 0;                     // stride0[47:32]=0 | stride1[15:0]=0
    g1[7] = 0;
    v4i z4 = {0, 0, 0, 0};         // groups 2/3: dims 2..4 unused
    v8i z8 = {0, 0, 0, 0, 0, 0, 0, 0};
    __builtin_amdgcn_tensor_load_to_lds(g0, g1, z4, z4, z8, 0);
}

// ---------------------------------------------------------------------------
// Projection GEMM: out = X[8192,1024] @ W[1024,1024] + bias, bf16 output.
// MODE 0: out[((b*H+h)*S+s)*64+d]   (Q, K head-major)
// MODE 1: out[((b*H+h)*64+d)*S+s]   (V transposed for P@V B-fragments)
// ---------------------------------------------------------------------------
template <int MODE>
__global__ __launch_bounds__(256) void gemm_proj_kernel(
    const float* __restrict__ X, const float* __restrict__ W,
    const float* __restrict__ bias, __bf16* __restrict__ out) {
    constexpr int K = CD, N = CD, SA = 40;
    __shared__ alignas(16) __bf16 As[128 * SA];
    __shared__ alignas(16) __bf16 Bs[128 * SA];

    const int t = threadIdx.x, lane = t & 31, wave = t >> 5;
    const int wm = wave >> 1, wn = wave & 1;
    const int n0blk = blockIdx.x * 128;
    const int m0blk = blockIdx.y * 128;

    v8f acc[2][4];
#pragma unroll
    for (int mi = 0; mi < 2; ++mi)
#pragma unroll
        for (int ni = 0; ni < 4; ++ni)
            acc[mi][ni] = v8f{0.f, 0.f, 0.f, 0.f, 0.f, 0.f, 0.f, 0.f};

    for (int kt = 0; kt < K; kt += 32) {
#pragma unroll
        for (int i = 0; i < 4; ++i) {
            int idx = t + 256 * i;
            int row = idx >> 3, c4 = (idx & 7) * 4;
            float4 x = *(const float4*)(X + (size_t)(m0blk + row) * K + kt + c4);
            union { __bf16 e[4]; uint2 u; } pk;
            pk.e[0] = (__bf16)x.x; pk.e[1] = (__bf16)x.y;
            pk.e[2] = (__bf16)x.z; pk.e[3] = (__bf16)x.w;
            *(uint2*)(As + row * SA + c4) = pk.u;
        }
#pragma unroll
        for (int i = 0; i < 4; ++i) {
            int idx = t + 256 * i;
            int kr = idx >> 5, nc = (idx & 31) * 4;
            float4 w = *(const float4*)(W + (size_t)(kt + kr) * N + n0blk + nc);
            Bs[(nc + 0) * SA + kr] = (__bf16)w.x;
            Bs[(nc + 1) * SA + kr] = (__bf16)w.y;
            Bs[(nc + 2) * SA + kr] = (__bf16)w.z;
            Bs[(nc + 3) * SA + kr] = (__bf16)w.w;
        }
        if (kt + 32 < K) {  // global_prefetch_b8 next k-tile
            __builtin_prefetch(X + (size_t)(m0blk + (t & 127)) * K +
                               (kt + 32) + ((t >> 7) << 4), 0, 0);
            __builtin_prefetch(W + (size_t)(kt + 32 + (t & 31)) * N +
                               n0blk + ((t >> 5) << 4), 0, 0);
        }
        __syncthreads();

        v16bf af0 = frag_ld(As, SA, wm * 32, 0, lane);
        v16bf af1 = frag_ld(As, SA, wm * 32 + 16, 0, lane);
#pragma unroll
        for (int ni = 0; ni < 4; ++ni) {
            v16bf bf = frag_ld(Bs, SA, wn * 64 + ni * 16, 0, lane);
            acc[0][ni] = wmma_bf16(af0, bf, acc[0][ni]);
            acc[1][ni] = wmma_bf16(af1, bf, acc[1][ni]);
        }
        __syncthreads();
    }

    const int half = lane >> 4, lr = lane & 15;
#pragma unroll
    for (int mi = 0; mi < 2; ++mi)
#pragma unroll
        for (int ni = 0; ni < 4; ++ni) {
            int mbase = m0blk + wm * 32 + mi * 16;
            int n = n0blk + wn * 64 + ni * 16 + lr;
            int h = n >> 6, d = n & 63;
            float bb = bias[n];
#pragma unroll
            for (int r = 0; r < 8; ++r) {
                int m = mbase + r + 8 * half;
                int b = m >> 11, s = m & (CS - 1);
                float v = acc[mi][ni][r] + bb;
                size_t o = (MODE == 0)
                    ? ((size_t)((b * CH + h) * CS + s)) * 64 + d
                    : ((size_t)((b * CH + h) * 64 + d)) * CS + s;
                out[o] = (__bf16)v;
            }
        }
}

// ---------------------------------------------------------------------------
// Fused attention: one workgroup per (b, h, 32-row block), 8 waves.
//  phase 1: scores = Q@K^T * 0.125 -> LDS P[32][2052] f32 (WMMA)
//  phase 2: coalesced mask apply + row softmax (wave32 shfl reductions),
//           coalesced f32 prob write to d_out
//  phase 3: context = P@V via WMMA (Vt gives contiguous B-frags), * 1/rowsum
// LDS 262,784 B dynamic (one WG per WGP).
// ---------------------------------------------------------------------------
__global__ __launch_bounds__(256) void attn_kernel(
    const __bf16* __restrict__ Q,   // [B,H,S,64]
    const __bf16* __restrict__ Kk,  // [B,H,S,64]
    const __bf16* __restrict__ Vt,  // [B,H,64,S]
    const unsigned char* __restrict__ mask,  // [B,S,S] bool
    float* __restrict__ attnOut,             // [B,H,S,S]
    __bf16* __restrict__ ctx) {              // [B,S,H*64]
    constexpr int PS = 2052;  // padded P row stride (floats)
    extern __shared__ char smem[];
    float* P    = (float*)smem;
    float* invs = (float*)(smem + (size_t)32 * PS * 4);

    const int blk = blockIdx.x;
    const int rb = blk & 63, h = (blk >> 6) & 15, b = blk >> 10;
    const int q0 = rb * 32;
    const int t = threadIdx.x, lane = t & 31, wave = t >> 5;
    const int half = lane >> 4, lr = lane & 15;

    const size_t headoff = ((size_t)(b * CH + h)) * CS * 64;
    const __bf16* Qh = Q + headoff;
    const __bf16* Kh = Kk + headoff;
    const __bf16* Vh = Vt + headoff;   // rows = d (64), stride S

    // ---- phase 1: raw scaled scores -------------------------------------
    v16bf aq[2][2];
#pragma unroll
    for (int mi = 0; mi < 2; ++mi)
#pragma unroll
        for (int ks = 0; ks < 2; ++ks)
            aq[mi][ks] = frag_ld(Qh, 64, q0 + mi * 16, ks * 32, lane);

    for (int nt = 0; nt < 16; ++nt) {
        int n0 = wave * 256 + nt * 16;
        v8f a0 = v8f{0.f,0.f,0.f,0.f,0.f,0.f,0.f,0.f};
        v8f a1 = v8f{0.f,0.f,0.f,0.f,0.f,0.f,0.f,0.f};
#pragma unroll
        for (int ks = 0; ks < 2; ++ks) {
            v16bf bk = frag_ld(Kh, 64, n0, ks * 32, lane);  // B = K^T tile
            a0 = wmma_bf16(aq[0][ks], bk, a0);
            a1 = wmma_bf16(aq[1][ks], bk, a1);
        }
        int col = n0 + lr;
#pragma unroll
        for (int r = 0; r < 8; ++r) {
            P[(r + 8 * half) * PS + col]      = a0[r] * 0.125f;
            P[(16 + r + 8 * half) * PS + col] = a1[r] * 0.125f;
        }
    }
    __syncthreads();

    // ---- phase 2: mask (coalesced) + softmax + write attention out -------
#pragma unroll
    for (int rr = 0; rr < 4; ++rr) {
        int row = wave * 4 + rr;
        float* pr = P + row * PS;
        const unsigned char* mrow =
            mask + ((size_t)b * CS + (q0 + row)) * CS;
        float mx = -3.0e38f;
        for (int i = 0; i < 64; ++i) {          // coalesced: 32 B per wave
            int c = lane + 32 * i;
            float v = mrow[c] ? -1e9f : pr[c];
            pr[c] = v;
            mx = fmaxf(mx, v);
        }
        mx = wave_max32(mx);
        float sum = 0.f;
        for (int i = 0; i < 64; ++i) {
            int c = lane + 32 * i;
            float e = __expf(pr[c] - mx);
            pr[c] = e;                           // keep unnormalized exp
            sum += e;
        }
        sum = wave_sum32(sum);
        float inv = 1.f / sum;
        if (lane == 0) invs[row] = inv;
        float* orow = attnOut +
            (((size_t)(b * CH + h)) * CS + (q0 + row)) * CS;
        for (int i = 0; i < 64; ++i) {
            int c = lane + 32 * i;
            orow[c] = pr[c] * inv;               // coalesced 128 B per wave
        }
    }
    __syncthreads();

    // ---- phase 3: context = P @ V ---------------------------------------
    const int mi = wave >> 2, ni = wave & 3;    // 2 x 4 tiles of 16x16
    const int m0 = mi * 16, d0 = ni * 16;
    v8f c = v8f{0.f,0.f,0.f,0.f,0.f,0.f,0.f,0.f};
    for (int k0 = 0; k0 < CS; k0 += 32) {
        v16bf ap = frag_ld_f32(P, PS, m0, k0, lane);
        v16bf bv = frag_ld(Vh, CS, d0, k0, lane);
        c = wmma_bf16(ap, bv, c);
    }
#pragma unroll
    for (int r = 0; r < 8; ++r) {
        int row = m0 + r + 8 * half;
        float v = c[r] * invs[row];
        int q = q0 + row, d = d0 + lr;
        ctx[((size_t)(b * CS + q)) * CD + h * 64 + d] = (__bf16)v;
    }
}

// ---------------------------------------------------------------------------
// Output projection: proj = ctx(bf16)[8192,1024] @ Wo[1024,1024] + bo, f32.
// A-tile staged by the Tensor Data Mover (wave 0 issues, TENSORcnt wait).
// ---------------------------------------------------------------------------
__global__ __launch_bounds__(256) void gemm_out_kernel(
    const __bf16* __restrict__ A, const float* __restrict__ W,
    const float* __restrict__ bias, float* __restrict__ out) {
    constexpr int K = CD, N = CD, SA = 40;
    __shared__ alignas(16) __bf16 As[128 * SA];
    __shared__ alignas(16) __bf16 Bs[128 * SA];

    const int t = threadIdx.x, lane = t & 31, wave = t >> 5;
    const int wm = wave >> 1, wn = wave & 1;
    const int n0blk = blockIdx.x * 128;
    const int m0blk = blockIdx.y * 128;
    const u32 as_off = (u32)(uintptr_t)(&As[0]);   // LDS offset = addr[31:0]

    v8f acc[2][4];
#pragma unroll
    for (int mi = 0; mi < 2; ++mi)
#pragma unroll
        for (int ni = 0; ni < 4; ++ni)
            acc[mi][ni] = v8f{0.f, 0.f, 0.f, 0.f, 0.f, 0.f, 0.f, 0.f};

    for (int kt = 0; kt < K; kt += 32) {
        if (wave == 0)  // TDM: 128 rows x 32 bf16, padded stride 40 in LDS
            tdm_load_tile(as_off, A + (size_t)m0blk * K + kt);
#pragma unroll
        for (int i = 0; i < 4; ++i) {
            int idx = t + 256 * i;
            int kr = idx >> 5, nc = (idx & 31) * 4;
            float4 w = *(const float4*)(W + (size_t)(kt + kr) * N + n0blk + nc);
            Bs[(nc + 0) * SA + kr] = (__bf16)w.x;
            Bs[(nc + 1) * SA + kr] = (__bf16)w.y;
            Bs[(nc + 2) * SA + kr] = (__bf16)w.z;
            Bs[(nc + 3) * SA + kr] = (__bf16)w.w;
        }
        if (wave == 0) __builtin_amdgcn_s_wait_tensorcnt(0);
        __syncthreads();

        v16bf af0 = frag_ld(As, SA, wm * 32, 0, lane);
        v16bf af1 = frag_ld(As, SA, wm * 32 + 16, 0, lane);
#pragma unroll
        for (int ni = 0; ni < 4; ++ni) {
            v16bf bf = frag_ld(Bs, SA, wn * 64 + ni * 16, 0, lane);
            acc[0][ni] = wmma_bf16(af0, bf, acc[0][ni]);
            acc[1][ni] = wmma_bf16(af1, bf, acc[1][ni]);
        }
        __syncthreads();
    }

    const int half = lane >> 4, lr = lane & 15;
#pragma unroll
    for (int mi = 0; mi < 2; ++mi)
#pragma unroll
        for (int ni = 0; ni < 4; ++ni) {
            int mbase = m0blk + wm * 32 + mi * 16;
            int n = n0blk + wn * 64 + ni * 16 + lr;
            float bb = bias[n];
#pragma unroll
            for (int r = 0; r < 8; ++r) {
                int m = mbase + r + 8 * half;
                out[(size_t)m * N + n] = acc[mi][ni][r] + bb;
            }
        }
}

// ---------------------------------------------------------------------------
// Residual add + LayerNorm: one workgroup per row of 1024.
// ---------------------------------------------------------------------------
__global__ __launch_bounds__(256) void ln_kernel(
    const float* __restrict__ proj, const float* __restrict__ resid,
    const float* __restrict__ gamma, const float* __restrict__ beta,
    float* __restrict__ out) {
    const int row = blockIdx.x;
    const int t = threadIdx.x, lane = t & 31, wave = t >> 5;
    __shared__ float redA[8], redB[8], stats[2];

    const float* pr = proj + (size_t)row * CD;
    const float* rr = resid + (size_t)row * CD;
    float x[4], s = 0.f, s2 = 0.f;
#pragma unroll
    for (int j = 0; j < 4; ++j) {
        int c = t + 256 * j;
        x[j] = pr[c] + rr[c];
        s += x[j];
        s2 += x[j] * x[j];
    }
    s = wave_sum32(s);
    s2 = wave_sum32(s2);
    if (lane == 0) { redA[wave] = s; redB[wave] = s2; }
    __syncthreads();
    if (t == 0) {
        float a = 0.f, b2 = 0.f;
#pragma unroll
        for (int w = 0; w < 8; ++w) { a += redA[w]; b2 += redB[w]; }
        float mean = a * (1.f / CD);
        float var = b2 * (1.f / CD) - mean * mean;
        stats[0] = mean;
        stats[1] = rsqrtf(var + 1e-5f);
    }
    __syncthreads();
    float mean = stats[0], rstd = stats[1];
#pragma unroll
    for (int j = 0; j < 4; ++j) {
        int c = t + 256 * j;
        out[(size_t)row * CD + c] = (x[j] - mean) * rstd * gamma[c] + beta[c];
    }
}

// ---------------------------------------------------------------------------
extern "C" void kernel_launch(void* const* d_in, const int* in_sizes, int n_in,
                              void* d_out, int out_size, void* d_ws,
                              size_t ws_size, hipStream_t stream) {
    (void)in_sizes; (void)n_in; (void)out_size; (void)ws_size;
    const float* Xq = (const float*)d_in[0];
    const float* Xk = (const float*)d_in[1];
    const float* Xv = (const float*)d_in[2];
    const unsigned char* mask = (const unsigned char*)d_in[3];  // jax bool
    const float* Wq = (const float*)d_in[4];
    const float* bq = (const float*)d_in[5];
    const float* Wk = (const float*)d_in[6];
    const float* bk = (const float*)d_in[7];
    const float* Wv = (const float*)d_in[8];
    const float* bv = (const float*)d_in[9];
    const float* Wo = (const float*)d_in[10];
    const float* bo = (const float*)d_in[11];
    const float* gamma = (const float*)d_in[12];
    const float* beta = (const float*)d_in[13];

    const size_t HALF = (size_t)CB * CH * CS * 64 * sizeof(__bf16);  // 16 MiB
    char* ws = (char*)d_ws;
    __bf16* Qb  = (__bf16*)(ws);
    __bf16* Kb  = (__bf16*)(ws + HALF);
    __bf16* Vtb = (__bf16*)(ws + 2 * HALF);
    __bf16* Ctx = (__bf16*)(ws + 3 * HALF);
    float*  Pj  = (float*)(ws + 4 * HALF);

    float* outAll = (float*)d_out;
    float* attnOut = outAll + (size_t)CB * CS * CD;  // output, then attention

    dim3 gGemm(CD / 128, (CB * CS) / 128);           // 8 x 64
    dim3 blk(256);

    gemm_proj_kernel<0><<<gGemm, blk, 0, stream>>>(Xq, Wq, bq, Qb);
    gemm_proj_kernel<0><<<gGemm, blk, 0, stream>>>(Xk, Wk, bk, Kb);
    gemm_proj_kernel<1><<<gGemm, blk, 0, stream>>>(Xv, Wv, bv, Vtb);

    size_t shb = (size_t)32 * 2052 * sizeof(float) + 32 * sizeof(float);
    attn_kernel<<<CB * CH * (CS / 32), blk, shb, stream>>>(
        Qb, Kb, Vtb, mask, attnOut, Ctx);

    gemm_out_kernel<<<gGemm, blk, 0, stream>>>(Ctx, Wo, bo, Pj);
    ln_kernel<<<CB * CS, blk, 0, stream>>>(Pj, Xq, gamma, beta, outAll);
}
